// Classifier_8418135900320
// MI455X (gfx1250) — matrix operand order repaired and checked by reading, not compiled, not run on hardware
//
#include <hip/hip_runtime.h>
#include <hip/hip_bf16.h>

// Problem constants (match reference setup_inputs: B=8192, D=512)
constexpr int BB = 8192;
constexpr int DD = 512;

typedef __attribute__((ext_vector_type(16))) __bf16 v16bf;
typedef __attribute__((ext_vector_type(8)))  __bf16 v8bf;
typedef __attribute__((ext_vector_type(8)))  float  v8f;

// ---------------------------------------------------------------------------
// Kernel 1: per-row prep. One block (128 threads) per row i:
//   - ||Z_i||, ||Y_i||, Z_i.Y_i in exact f32
//   - bf16 copies of both rows (for WMMA)
//   - invx[i] = 1/||Z_i||
//   - thresh[i] = (Z_i.Y_i / max(||Z_i||*||Y_i||, eps)) * ||Y_i||
//     so that the epilogue test  dot*invy*invx > diag  reduces to
//     dot*invx > thresh (invy > 0), removing one load+mul per element.
//   - zero the per-row "greater" counter
// ---------------------------------------------------------------------------
__global__ __launch_bounds__(128) void prep_kernel(
    const float* __restrict__ Z, const float* __restrict__ Y,
    __bf16* __restrict__ Xbf, __bf16* __restrict__ Ybf,
    float* __restrict__ invx, float* __restrict__ thresh,
    unsigned* __restrict__ counts) {
  __shared__ float s0[128], s1[128], s2[128];
  const int row = blockIdx.x;
  const int t = threadIdx.x;

  const float4 z = ((const float4*)(Z + (size_t)row * DD))[t];
  const float4 y = ((const float4*)(Y + (size_t)row * DD))[t];

  float zz = z.x * z.x + z.y * z.y + z.z * z.z + z.w * z.w;
  float yy = y.x * y.x + y.y * y.y + y.z * y.z + y.w * y.w;
  float zy = z.x * y.x + z.y * y.y + z.z * y.z + z.w * y.w;

  s0[t] = zz; s1[t] = yy; s2[t] = zy;
  __syncthreads();
  for (int off = 64; off > 0; off >>= 1) {
    if (t < off) { s0[t] += s0[t + off]; s1[t] += s1[t + off]; s2[t] += s2[t + off]; }
    __syncthreads();
  }

  // bf16 conversion (RNE via __bf16 arithmetic conversion)
  const size_t base = (size_t)row * DD + (size_t)t * 4;
  Xbf[base + 0] = (__bf16)z.x; Xbf[base + 1] = (__bf16)z.y;
  Xbf[base + 2] = (__bf16)z.z; Xbf[base + 3] = (__bf16)z.w;
  Ybf[base + 0] = (__bf16)y.x; Ybf[base + 1] = (__bf16)y.y;
  Ybf[base + 2] = (__bf16)y.z; Ybf[base + 3] = (__bf16)y.w;

  if (t == 0) {
    const float xn = sqrtf(s0[0]);
    const float yn = sqrtf(s1[0]);
    invx[row] = 1.0f / xn;
    thresh[row] = (s2[0] / fmaxf(xn * yn, 1e-8f)) * yn;
    counts[row] = 0u;
  }
}

// ---------------------------------------------------------------------------
// WMMA tile loaders (row-major source, stride DD bf16 elements).
// A 16x32 bf16 layout: lanes 0-15 (M=lane) hold K=0..7 then 16..23;
//                      lanes 16-31 hold K=8..15 then 24..31.
// B 32x16 bf16 layout: lanes 0-15 (N=lane) hold K=0..7 then 16..23;
//                      lanes 16-31 hold K=8..15 then 24..31.
// Both map to two 16-byte global_load_b128 per lane.
// ---------------------------------------------------------------------------
__device__ __forceinline__ v16bf load_a_tile(const __bf16* __restrict__ base,
                                             int row, int k, unsigned lane) {
  const __bf16* p = base + (size_t)row * DD + k + ((lane & 16u) ? 8 : 0);
  const v8bf c0 = *(const v8bf*)(p);
  const v8bf c1 = *(const v8bf*)(p + 16);
  v16bf r;
#pragma unroll
  for (int i = 0; i < 8; ++i) { r[i] = c0[i]; r[i + 8] = c1[i]; }
  return r;
}

__device__ __forceinline__ v16bf load_b_tile(const __bf16* __restrict__ base,
                                             int col, int k, unsigned lane) {
  const __bf16* p = base + (size_t)col * DD + k + ((lane & 16u) ? 8 : 0);
  const v8bf c0 = *(const v8bf*)(p);
  const v8bf c1 = *(const v8bf*)(p + 16);
  v16bf r;
#pragma unroll
  for (int i = 0; i < 8; ++i) { r[i] = c0[i]; r[i + 8] = c1[i]; }
  return r;
}

// ---------------------------------------------------------------------------
// Kernel 2: S = Ybf @ Xbf^T via v_wmma_f32_16x16x32_bf16.
// Block = 256 threads = 8 waves; block tile 128(M) x 128(N).
// Wave tile 32(M) x 64(N) = 2x4 of 16x16: 8 WMMAs per K-step fed by only
// 6 tile loads (2 A + 4 B) -> load/WMMA ratio 0.75; unique-per-block traffic
// 256KB over 4096 blocks (~1 GB of L2 reads) stays under the L2 ceiling.
//
// Epilogue (lean): pred = (c * invx[col] > thresh[row]) && (col != row),
// accumulated into packed byte counters (one byte per row v, max 4 per byte
// over the 4 N-subtiles), reduced across the 16 column lanes with 4
// __shfl_xor steps (masks 1,2,4,8 stay within each 16-lane half, which hold
// different rows), then 32 parallel per-row atomics per wave.
// ---------------------------------------------------------------------------
__global__ __launch_bounds__(256) void gemm_count_kernel(
    const __bf16* __restrict__ Ybf, const __bf16* __restrict__ Xbf,
    const float* __restrict__ invx, const float* __restrict__ thresh,
    unsigned* __restrict__ counts) {
  const unsigned tid = threadIdx.x;
  const unsigned lane = tid & 31u;
  const unsigned wave = tid >> 5;

  const int rm0 = (int)blockIdx.y * 128 + (int)(wave & 3u) * 32;  // M: Y rows
  const int cn0 = (int)blockIdx.x * 128 + (int)(wave >> 2) * 64;  // N: X rows

  v8f acc[2][4];
#pragma unroll
  for (int m = 0; m < 2; ++m)
#pragma unroll
    for (int n = 0; n < 4; ++n) acc[m][n] = {};

  const int arow0 = rm0 + (int)(lane & 15u);
  const int bcol0 = cn0 + (int)(lane & 15u);

  for (int k = 0; k < DD; k += 32) {
    v16bf a[2], b[4];
#pragma unroll
    for (int m = 0; m < 2; ++m) a[m] = load_a_tile(Ybf, arow0 + m * 16, k, lane);
#pragma unroll
    for (int n = 0; n < 4; ++n) b[n] = load_b_tile(Xbf, bcol0 + n * 16, k, lane);
#pragma unroll
    for (int m = 0; m < 2; ++m)
#pragma unroll
      for (int n = 0; n < 4; ++n)
        acc[m][n] = __builtin_amdgcn_wmma_f32_16x16x32_bf16(
            false, a[m], false, b[n], (short)0, acc[m][n], false, false);
  }

  // ---- epilogue ----
  const int hoff = (lane & 16u) ? 8 : 0;  // C layout: upper half-wave = rows +8

  // Per-lane row thresholds: rows rbase(m)..rbase(m)+7, two b128 loads per m.
  float th[2][8];
#pragma unroll
  for (int m = 0; m < 2; ++m) {
    const int rbase = rm0 + m * 16 + hoff;  // multiple of 8 -> 32B aligned
    *(float4*)&th[m][0] = *(const float4*)(thresh + rbase);
    *(float4*)&th[m][4] = *(const float4*)(thresh + rbase + 4);
  }

  // Packed byte counters: pk[m][0] holds rows v=0..3, pk[m][1] rows v=4..7.
  unsigned pk[2][2] = {{0u, 0u}, {0u, 0u}};
#pragma unroll
  for (int m = 0; m < 2; ++m) {
    const int rbase = rm0 + m * 16 + hoff;
#pragma unroll
    for (int n = 0; n < 4; ++n) {
      const int col = cn0 + n * 16 + (int)(lane & 15u);
      const float ix = invx[col];
#pragma unroll
      for (int v = 0; v < 8; ++v) {
        const bool pred = (acc[m][n][v] * ix > th[m][v]) && (col != rbase + v);
        pk[m][v >> 2] += ((unsigned)pred) << ((v & 3) * 8);
      }
    }
  }

  // Reduce across the 16 column lanes of each half-wave (bytes <= 64, no carry).
#pragma unroll
  for (int d = 1; d < 16; d <<= 1) {
#pragma unroll
    for (int m = 0; m < 2; ++m) {
      pk[m][0] += __shfl_xor(pk[m][0], d, 32);
      pk[m][1] += __shfl_xor(pk[m][1], d, 32);
    }
  }

  // Lanes 0..7 own rows rbase+0..7 of half 0; lanes 16..23 rows of half 1.
  if ((lane & 15u) < 8u) {
    const int r = (int)(lane & 7u);
#pragma unroll
    for (int m = 0; m < 2; ++m) {
      const unsigned word = (r & 4) ? pk[m][1] : pk[m][0];
      const unsigned c = (word >> ((r & 3) * 8)) & 0xffu;
      atomicAdd(&counts[rm0 + m * 16 + hoff + r], c);
    }
  }
}

// ---------------------------------------------------------------------------
// Kernel 3: counts -> (top1, top10).
// top1[i]  hit iff no element strictly greater than diagonal  (count == 0)
// top10[i] hit iff fewer than 10 strictly greater             (count < 10)
// ---------------------------------------------------------------------------
__global__ __launch_bounds__(256) void finalize_kernel(
    const unsigned* __restrict__ counts, float* __restrict__ out) {
  __shared__ unsigned sh1[256], sh10[256];
  const unsigned t = threadIdx.x;
  unsigned a1 = 0, a10 = 0;
  for (int i = (int)t; i < BB; i += 256) {
    const unsigned c = counts[i];
    a1 += (c == 0u);
    a10 += (c < 10u);
  }
  sh1[t] = a1; sh10[t] = a10;
  __syncthreads();
  for (int off = 128; off > 0; off >>= 1) {
    if (t < off) { sh1[t] += sh1[t + off]; sh10[t] += sh10[t + off]; }
    __syncthreads();
  }
  if (t == 0) {
    out[0] = (float)sh1[0] / (float)BB;
    out[1] = (float)sh10[0] / (float)BB;
  }
}

// ---------------------------------------------------------------------------
extern "C" void kernel_launch(void* const* d_in, const int* in_sizes, int n_in,
                              void* d_out, int out_size, void* d_ws, size_t ws_size,
                              hipStream_t stream) {
  (void)in_sizes; (void)n_in; (void)out_size; (void)ws_size;
  const float* Z = (const float*)d_in[0];  // "x" in reference
  const float* Y = (const float*)d_in[1];  // "y" in reference
  float* out = (float*)d_out;

  char* w = (char*)d_ws;
  __bf16* Ybf = (__bf16*)w;                                      // 8 MB
  __bf16* Xbf = (__bf16*)(w + (size_t)BB * DD * sizeof(__bf16)); // 8 MB
  float* invx = (float*)(w + (size_t)2 * BB * DD * sizeof(__bf16));
  float* thresh = invx + BB;
  unsigned* counts = (unsigned*)(thresh + BB);

  prep_kernel<<<BB, 128, 0, stream>>>(Z, Y, Xbf, Ybf, invx, thresh, counts);

  dim3 grid(BB / 128, BB / 128);  // 64 x 64 blocks of 128x128 tiles
  gemm_count_kernel<<<grid, 256, 0, stream>>>(Ybf, Xbf, invx, thresh, counts);

  finalize_kernel<<<1, 256, 0, stream>>>(counts, out);
}
